// Qwen3MoeSparseMoeBlock_3753801417653
// MI455X (gfx1250) — matrix operand order repaired and checked by reading, not compiled, not run on hardware
//
#include <hip/hip_runtime.h>
#include <hip/hip_bf16.h>
#include <cstdint>

#define H_DIM 2048
#define I_DIM 768
#define E_NUM 32
#define TOPK  8
#define T_NUM 512
#define TTILE 32
#define NCHUNK 512

typedef __attribute__((ext_vector_type(16))) __bf16 v16bf;
typedef __attribute__((ext_vector_type(8)))  __bf16 v8bf;
typedef __attribute__((ext_vector_type(8)))  float  v8f;

// pack two f32 -> two bf16 in one dword (RNE via native fptrunc)
__device__ __forceinline__ uint32_t pk2(float lo, float hi) {
  union { __bf16 b[2]; uint32_t u; } o;
  o.b[0] = (__bf16)lo;
  o.b[1] = (__bf16)hi;
  return o.u;
}

__device__ __forceinline__ float wave_max(float v) {
  #pragma unroll
  for (int off = 16; off > 0; off >>= 1) v = fmaxf(v, __shfl_xor(v, off, 32));
  return v;
}
__device__ __forceinline__ float wave_sum(float v) {
  #pragma unroll
  for (int off = 16; off > 0; off >>= 1) v += __shfl_xor(v, off, 32);
  return v;
}

__device__ __forceinline__ v16bf pack16(v8bf lo, v8bf hi) {
  v16bf r;
  #pragma unroll
  for (int i = 0; i < 8; ++i) { r[i] = lo[i]; r[i + 8] = hi[i]; }
  return r;
}
// A fragment (16x32 bf16): lane m = lane&15, two 16B chunks at k = half*8 and 16+half*8
__device__ __forceinline__ v16bf load_a(const __bf16* rowbase, int half) {
  const v8bf* p0 = (const v8bf*)(rowbase + half * 8);
  const v8bf* p1 = (const v8bf*)(rowbase + 16 + half * 8);
  return pack16(*p0, *p1);
}
// B fragment (32x16 bf16) from pre-transposed tile: lane n = lane&15,
// two 16B chunks at k = half*8 and 16+half*8
__device__ __forceinline__ v16bf load_b(const __bf16* rowbase, int half) {
  const v8bf* p0 = (const v8bf*)(rowbase + half * 8);
  const v8bf* p1 = (const v8bf*)(rowbase + 16 + half * 8);
  return pack16(*p0, *p1);
}

// -------------------- zero output + dispatch counters --------------------
__global__ void zero_kernel(float* __restrict__ y, int n, int* __restrict__ cnt) {
  int i = blockIdx.x * blockDim.x + threadIdx.x;
  if (i < n) y[i] = 0.0f;
  if (i < E_NUM) cnt[i] = 0;
}

// ------ router: softmax + top-8 + renorm + per-expert token dispatch ------
__global__ void router_kernel(const float* __restrict__ x,
                              const float* __restrict__ gw,
                              int* __restrict__ cnt,
                              int* __restrict__ idxb,
                              float* __restrict__ wgtb) {
  int lane = threadIdx.x & 31;   // expert id (E == 32 == wave32)
  int wv   = threadIdx.x >> 5;
  int t    = blockIdx.x * 8 + wv;

  const float4* xv = (const float4*)(x + (size_t)t * H_DIM);
  const float4* gv = (const float4*)(gw + (size_t)lane * H_DIM);
  float acc = 0.f;
  for (int i = 0; i < H_DIM / 4; ++i) {
    float4 a = xv[i], b = gv[i];
    acc += a.x * b.x + a.y * b.y + a.z * b.z + a.w * b.w;
  }
  float m = wave_max(acc);
  float p = __expf(acc - m);
  float s = wave_sum(p);
  float prob = p / s;

  float cur = prob, topsum = 0.f;
  bool sel = false;
  #pragma unroll
  for (int i = 0; i < TOPK; ++i) {
    float mx = wave_max(cur);
    unsigned long long b = __ballot(cur == mx);
    int win = __builtin_ctzll(b);
    if (lane == win) { sel = true; cur = -1.0f; }
    topsum += mx;
  }
  if (sel) {
    int pos = atomicAdd(&cnt[lane], 1);          // compact token list per expert
    idxb[lane * T_NUM + pos] = t;
    wgtb[lane * T_NUM + pos] = prob / topsum;
  }
}

// -------------------- expert GLU MLP + combine (gathered tokens) --------------------
// LDS layout (bytes), regions overlaid across stages:
//   stage1: Xs[32][40]bf16 @0 (2560) | Wg[768][40]bf16 @2560 (61440) | Wu @64000 (61440)
//   stage2/3: GLU[32][776]bf16 @0 (49664) | Wd[512][40]bf16 @49664 (40960)
//   comb cache: 32 f32 @125440 | token ids: 32 i32 @125568
#define XS_OFF   0
#define WG_OFF   2560
#define WU_OFF   (WG_OFF + 61440)
#define GLU_OFF  0
#define WD_OFF   49664
#define COMB_OFF 125440
#define TOK_OFF  125568
#define SMEM_BYTES 125696

__launch_bounds__(256, 1)
__global__ void moe_expert_kernel(const float* __restrict__ x,
                                  const float* __restrict__ gp,
                                  const float* __restrict__ up,
                                  const float* __restrict__ dp,
                                  const int* __restrict__ cnt,
                                  const int* __restrict__ idxb,
                                  const float* __restrict__ wgtb,
                                  float* __restrict__ y) {
  const int e = blockIdx.y;
  const int cntE = cnt[e];
  if ((int)blockIdx.x * TTILE >= cntE) return;   // uniform early-exit: dead tile

  __shared__ __align__(16) char smem[SMEM_BYTES];
  __bf16* Xs  = (__bf16*)(smem + XS_OFF);
  __bf16* Wg  = (__bf16*)(smem + WG_OFF);
  __bf16* Wu  = (__bf16*)(smem + WU_OFF);
  __bf16* GLU = (__bf16*)(smem + GLU_OFF);
  __bf16* Wd  = (__bf16*)(smem + WD_OFF);
  float*  cmb = (float*)(smem + COMB_OFF);
  int*    tks = (int*)(smem + TOK_OFF);

  const int tid  = threadIdx.x;
  const int lane = tid & 31;
  const int wv   = tid >> 5;    // 8 waves
  const int mt   = wv & 1;      // M-tile (16 tokens each)
  const int wq   = wv >> 1;     // 0..3
  const int half = lane >> 4;
  const int l16  = lane & 15;

  // gather this tile's token rows (pad with token 0 / weight 0)
  if (tid < TTILE) {
    int gi = blockIdx.x * TTILE + tid;
    bool valid = gi < cntE;
    tks[tid] = valid ? idxb[e * T_NUM + gi] : 0;
    cmb[tid] = valid ? wgtb[e * T_NUM + gi] : 0.f;
  }
  __syncthreads();

  const size_t wbase = (size_t)e * H_DIM * I_DIM;
  const size_t dbase = (size_t)e * I_DIM * H_DIM;

  // ---- stage 1: G = X@Wg, U = X@Wu  ([32,2048]x[2048,768]) ----
  v8f accg[12], accu[12];
  #pragma unroll
  for (int j = 0; j < 12; ++j) { accg[j] = {}; accu[j] = {}; }

  for (int h0 = 0; h0 < H_DIM; h0 += 32) {
    // X tile: 32 gathered rows x 16 k-pairs, float2 loads -> packed b32 LDS stores
    #pragma unroll
    for (int i = 0; i < 2; ++i) {
      int idx = tid + i * 256;
      int tok = idx >> 4, kp = idx & 15;
      const float2 v2 = *(const float2*)(x + (size_t)tks[tok] * H_DIM + h0 + 2 * kp);
      *(uint32_t*)(Xs + tok * 40 + 2 * kp) = pk2(v2.x, v2.y);
    }
    // W tiles, staged transposed [n][k]; pairs along k (two rows) and n (float2)
    for (int kp = 0; kp < 16; ++kp) {
      const float* g0 = gp + wbase + (size_t)(h0 + 2 * kp) * I_DIM;
      const float* u0 = up + wbase + (size_t)(h0 + 2 * kp) * I_DIM;
      auto stage = [&](int cp) {
        float2 ga = *(const float2*)(g0 + 2 * cp);
        float2 gb = *(const float2*)(g0 + I_DIM + 2 * cp);
        float2 ua = *(const float2*)(u0 + 2 * cp);
        float2 ub = *(const float2*)(u0 + I_DIM + 2 * cp);
        *(uint32_t*)(Wg + (2 * cp)     * 40 + 2 * kp) = pk2(ga.x, gb.x);
        *(uint32_t*)(Wg + (2 * cp + 1) * 40 + 2 * kp) = pk2(ga.y, gb.y);
        *(uint32_t*)(Wu + (2 * cp)     * 40 + 2 * kp) = pk2(ua.x, ub.x);
        *(uint32_t*)(Wu + (2 * cp + 1) * 40 + 2 * kp) = pk2(ua.y, ub.y);
      };
      stage(tid);                       // col-pairs 0..255
      if (tid < 128) stage(256 + tid);  // col-pairs 256..383
    }
    __syncthreads();

    // prefetch next k-step's weight tiles into cache while we compute
    if (h0 + 32 < H_DIM) {
      const char* pg = (const char*)(gp + wbase + (size_t)(h0 + 32) * I_DIM);
      const char* pu = (const char*)(up + wbase + (size_t)(h0 + 32) * I_DIM);
      #pragma unroll
      for (int i = 0; i < 3; ++i) {
        __builtin_prefetch(pg + (size_t)(tid + i * 256) * 128, 0, 3);
        __builtin_prefetch(pu + (size_t)(tid + i * 256) * 128, 0, 3);
      }
    }

    v16bf a = load_a(Xs + (mt * 16 + l16) * 40, half);
    #pragma unroll
    for (int j = 0; j < 12; ++j) {
      int ncol = (wq * 12 + j) * 16 + l16;
      v16bf bg = load_b(Wg + ncol * 40, half);
      v16bf bu = load_b(Wu + ncol * 40, half);
      accg[j] = __builtin_amdgcn_wmma_f32_16x16x32_bf16(false, a, false, bg,
                                                        (short)0, accg[j], false, false);
      accu[j] = __builtin_amdgcn_wmma_f32_16x16x32_bf16(false, a, false, bu,
                                                        (short)0, accu[j], false, false);
    }
    __syncthreads();
  }

  // ---- stage 2: GLU = silu(G) * U -> LDS (bf16) ----
  #pragma unroll
  for (int j = 0; j < 12; ++j) {
    int nb = (wq * 12 + j) * 16 + l16;
    #pragma unroll
    for (int r = 0; r < 8; ++r) {
      float g = accg[j][r], u = accu[j][r];
      float v = (g / (1.f + __expf(-g))) * u;
      int tok = mt * 16 + half * 8 + r;
      GLU[tok * 776 + nb] = (__bf16)v;
    }
  }
  __syncthreads();

  // ---- stage 3: O = GLU @ Wd ([32,768]x[768,2048]), weighted atomic combine ----
  for (int nc0 = 0; nc0 < H_DIM; nc0 += NCHUNK) {
    v8f acc[8];
    #pragma unroll
    for (int j = 0; j < 8; ++j) acc[j] = {};
    for (int k0 = 0; k0 < I_DIM; k0 += 32) {
      // stage Wd tile transposed [n=512][k=32]: 16 k-pairs x 256 col-pairs
      const float* d0 = dp + dbase + (size_t)k0 * H_DIM + nc0;
      #pragma unroll
      for (int i = 0; i < 16; ++i) {
        int uidx = tid + i * 256;     // 0..4095
        int kp = uidx >> 8, cp = uidx & 255;
        const float* b0 = d0 + (size_t)(2 * kp) * H_DIM + 2 * cp;
        float2 r0 = *(const float2*)(b0);
        float2 r1 = *(const float2*)(b0 + H_DIM);
        *(uint32_t*)(Wd + (2 * cp)     * 40 + 2 * kp) = pk2(r0.x, r1.x);
        *(uint32_t*)(Wd + (2 * cp + 1) * 40 + 2 * kp) = pk2(r0.y, r1.y);
      }
      __syncthreads();

      // prefetch next k-slab of Wd (32 rows x 2KB = 512 lines of 128B)
      if (k0 + 32 < I_DIM) {
        const char* pd = (const char*)(dp + dbase + (size_t)(k0 + 32) * H_DIM + nc0);
        #pragma unroll
        for (int i = 0; i < 2; ++i) {
          int li = tid + i * 256;     // row = li>>4, line-in-row = li&15
          __builtin_prefetch(pd + (size_t)(li >> 4) * (H_DIM * 4) +
                                  (size_t)(li & 15) * 128, 0, 3);
        }
      }

      v16bf a = load_a(GLU + (mt * 16 + l16) * 776 + k0, half);
      #pragma unroll
      for (int j = 0; j < 8; ++j) {
        v16bf b = load_b(Wd + ((wq * 8 + j) * 16 + l16) * 40, half);
        acc[j] = __builtin_amdgcn_wmma_f32_16x16x32_bf16(false, a, false, b,
                                                         (short)0, acc[j], false, false);
      }
      __syncthreads();
    }
    #pragma unroll
    for (int j = 0; j < 8; ++j) {
      int ncol = nc0 + (wq * 8 + j) * 16 + l16;
      #pragma unroll
      for (int r = 0; r < 8; ++r) {
        int tok = mt * 16 + half * 8 + r;
        float cw = cmb[tok];
        if (cw != 0.f)
          atomicAdd(&y[(size_t)tks[tok] * H_DIM + ncol], cw * acc[j][r]);
      }
    }
  }
}

extern "C" void kernel_launch(void* const* d_in, const int* in_sizes, int n_in,
                              void* d_out, int out_size, void* d_ws, size_t ws_size,
                              hipStream_t stream) {
  const float* x  = (const float*)d_in[0];
  const float* gw = (const float*)d_in[1];
  const float* gp = (const float*)d_in[2];
  const float* up = (const float*)d_in[3];
  const float* dp = (const float*)d_in[4];
  float* y = (float*)d_out;

  // workspace: cnt[32] @0 | idx[32*512] @128 | wgt[32*512] @65664
  int*   cnt  = (int*)d_ws;
  int*   idxb = (int*)((char*)d_ws + 128);
  float* wgtb = (float*)((char*)d_ws + 128 + E_NUM * T_NUM * 4);

  int n = T_NUM * H_DIM;
  zero_kernel<<<(n + 255) / 256, 256, 0, stream>>>(y, n, cnt);
  router_kernel<<<T_NUM / 8, 256, 0, stream>>>(x, gw, cnt, idxb, wgtb);
  dim3 grid(T_NUM / TTILE, E_NUM);
  moe_expert_kernel<<<grid, 256, 0, stream>>>(x, gp, up, dp, cnt, idxb, wgtb, y);
}
